// CombineAttention_69630009803019
// MI455X (gfx1250) — matrix operand order repaired and checked by reading, not compiled, not run on hardware
//
#include <hip/hip_runtime.h>

typedef __attribute__((ext_vector_type(16))) __bf16 v16bf;
typedef __attribute__((ext_vector_type(8)))  __bf16 v8bf;
typedef __attribute__((ext_vector_type(4)))  __bf16 v4bf;
typedef __attribute__((ext_vector_type(8)))  float  v8f;

#define BATCH 2
#define SEQ_T 4096
#define SEQ_S 1024
#define CH    1024
#define NH    16
#define HD    64
// ln(4096) / sqrt(64)
#define QSCALE 1.03972077083992f

// ---------------------------------------------------------------------------
// WMMA helpers (CDNA5 16x16x32 bf16 -> f32)
// ---------------------------------------------------------------------------
static __device__ __forceinline__ v8f wmma_bf16(v16bf a, v16bf b, v8f c) {
  return __builtin_amdgcn_wmma_f32_16x16x32_bf16(
      /*neg_a=*/false, a, /*neg_b=*/false, b,
      /*c_mod=*/(short)0, c, /*reuse_a=*/false, /*reuse_b=*/false);
}

// A operand fragment (16 rows x 32 K) from row-major [rows, ld] bf16.
// Lanes 0-15 hold K 0..15; lanes 16-31 hold K 16..31.
static __device__ __forceinline__ v16bf load_frag_a(const __bf16* p, int ld) {
  const int lane = threadIdx.x & 31;
  const int r = lane & 15, hf = lane >> 4;
  const __bf16* q = p + (size_t)r * ld + hf * 16;
  return *(const v16bf*)q;           // K = hf*16 + 0..15
}

// B operand fragment (32 K x 16 cols), B stored row-major as [cols, ld].
// Lane holds column lane&15; lanes 0-15: K 0..15, lanes 16-31: K 16..31.
static __device__ __forceinline__ v16bf load_frag_b(const __bf16* p, int ld) {
  const int lane = threadIdx.x & 31;
  const int n = lane & 15, hf = lane >> 4;
  return *(const v16bf*)(p + (size_t)n * ld + hf * 16);
}

// ---------------------------------------------------------------------------
// f32 -> bf16 conversion (vectorized, grid-stride)
// ---------------------------------------------------------------------------
__global__ void __launch_bounds__(256)
cvt_f32_bf16(const float* __restrict__ in, __bf16* __restrict__ out, int n4) {
  int i = blockIdx.x * blockDim.x + threadIdx.x;
  const int stride = gridDim.x * blockDim.x;
  for (; i < n4; i += stride) {
    float4 f = ((const float4*)in)[i];
    v4bf o;
    o[0] = (__bf16)f.x; o[1] = (__bf16)f.y; o[2] = (__bf16)f.z; o[3] = (__bf16)f.w;
    ((v4bf*)out)[i] = o;
  }
}

// ---------------------------------------------------------------------------
// GEMM: Out[m,n] = sum_k A[m,k] * Bw[n,k]   (A:[M,K] bf16, Bw:[N,K] bf16)
// One wave computes a 32(M) x 64(N) tile; software-pipelined k-loop,
// ping-pong unrolled x2 so the fragment rotation is renamed away.
// MODE 0: bf16 row-major out
// MODE 1: bf16 row-major out, scaled by QSCALE*qm[n%64]   (q projection)
// MODE 2: bf16 transposed-per-head out vT[b,h,d,t]        (v projection)
// MODE 3: f32 row-major out                               (final projection)
// ---------------------------------------------------------------------------
template <int MODE>
__global__ void __launch_bounds__(256)
gemm_bf16_kernel(const __bf16* __restrict__ A, const __bf16* __restrict__ Bw,
                 void* __restrict__ Out, const float* __restrict__ qm,
                 int M, int N, int K) {
  const int wavesN = N >> 6;
  const int wave = blockIdx.x * 8 + (threadIdx.x >> 5);
  const int wm = wave / wavesN, wn = wave % wavesN;
  const int m0 = wm * 32, n0 = wn * 64;
  if (m0 >= M) return;
  const int lane = threadIdx.x & 31;
  const int r15 = lane & 15, hf = lane >> 4;

  v8f acc[2][4] = {};

  // pipeline prologue: fragments for k0 = 0
  v16bf a0 = load_frag_a(A + (size_t)m0 * K, K);
  v16bf a1 = load_frag_a(A + (size_t)(m0 + 16) * K, K);
  v16bf bfr[4];
#pragma unroll
  for (int j = 0; j < 4; j++) bfr[j] = load_frag_b(Bw + (size_t)(n0 + 16 * j) * K, K);

  // K is a multiple of 64 -> trip count is even; unroll x2 so the
  // "next -> current" fragment rotation becomes pure SSA renaming.
#pragma unroll 2
  for (int k0 = 0; k0 < K; k0 += 32) {
    // L2 prefetch ~4 k-steps ahead (one row per lane; speculative => safe OOB)
    __builtin_prefetch(A  + (size_t)(m0 + lane) * K + k0 + 128, 0, 1);
    __builtin_prefetch(Bw + (size_t)(n0 + lane) * K + k0 + 128, 0, 1);
    __builtin_prefetch(Bw + (size_t)(n0 + 32 + lane) * K + k0 + 128, 0, 1);

    // issue next iteration's fragment loads (branchless wrap on last step)
    const int kn = (k0 + 32 < K) ? (k0 + 32) : 0;
    v16bf na0 = load_frag_a(A + (size_t)m0 * K + kn, K);
    v16bf na1 = load_frag_a(A + (size_t)(m0 + 16) * K + kn, K);
    v16bf nb[4];
#pragma unroll
    for (int j = 0; j < 4; j++)
      nb[j] = load_frag_b(Bw + (size_t)(n0 + 16 * j) * K + kn, K);

    // compute on current fragments (loaded one iteration ago)
#pragma unroll
    for (int j = 0; j < 4; j++) {
      acc[0][j] = wmma_bf16(a0, bfr[j], acc[0][j]);
      acc[1][j] = wmma_bf16(a1, bfr[j], acc[1][j]);
    }

    a0 = na0; a1 = na1;
#pragma unroll
    for (int j = 0; j < 4; j++) bfr[j] = nb[j];
  }

#pragma unroll
  for (int i = 0; i < 2; i++) {
#pragma unroll
    for (int j = 0; j < 4; j++) {
      const int col = n0 + 16 * j + r15;
      if (MODE == 2) {
        // v transposed store: vT[b, h, d, t]; rows (t) contiguous per lane
        const int hh = col >> 6, d = col & 63;
        const int rowBase = m0 + 16 * i + 8 * hf;
        const int bb = rowBase / SEQ_T;
        const int t  = rowBase - bb * SEQ_T;
        v8bf pk;
#pragma unroll
        for (int r = 0; r < 8; r++) pk[r] = (__bf16)acc[i][j][r];
        *(v8bf*)((__bf16*)Out + (((size_t)bb * NH + hh) * HD + d) * SEQ_T + t) = pk;
      } else {
#pragma unroll
        for (int r = 0; r < 8; r++) {
          const int row = m0 + 16 * i + r + 8 * hf;
          float v = acc[i][j][r];
          if (MODE == 1) v *= QSCALE * qm[col & 63];
          if (MODE == 3) ((float*)Out)[(size_t)row * N + col] = v;
          else           ((__bf16*)Out)[(size_t)row * N + col] = (__bf16)v;
        }
      }
    }
  }
}

// ---------------------------------------------------------------------------
// Flash attention core.  One wave handles one (b, h, 16-query tile).
// q already scaled by ln(T)*qm/sqrt(D).  Monotonic mask: query i attends
// keys <= 4*i + 3 (T/sT == 4).  K/V fragment loads are pipelined so the
// softmax VALU work hides global-load latency; chunk loop is ping-pong
// unrolled x2 (trip count is always even: 2*it+2 causal, 128 full).
// ---------------------------------------------------------------------------
__global__ void __launch_bounds__(256)
attn_kernel(const __bf16* __restrict__ qb, const __bf16* __restrict__ kb,
            const __bf16* __restrict__ vt, __bf16* __restrict__ yb,
            const int* __restrict__ causalp) {
  __shared__ __bf16 Plds[8 * 16 * 32];  // per-wave 16x32 probability tile

  const int lane = threadIdx.x & 31;
  const int r15 = lane & 15, hf = lane >> 4;
  const int wave = blockIdx.x * 8 + (threadIdx.x >> 5);
  const int it = wave & 63;            // SEQ_S/16 == 64 query tiles
  const int bh = wave >> 6;
  const int b = bh / NH, h = bh % NH;
  const int qi0 = it * 16;
  const int causal = *causalp;

  const __bf16* qrow = qb + ((size_t)b * SEQ_S + qi0) * CH + h * HD;
  v16bf qa0 = load_frag_a(qrow, CH);        // d 0..31
  v16bf qa1 = load_frag_a(qrow + 32, CH);   // d 32..63

  v8f y0 = {}, y1 = {}, y2 = {}, y3 = {};
  float mrun[8], lrun[8];
#pragma unroll
  for (int r = 0; r < 8; r++) { mrun[r] = -1e30f; lrun[r] = 0.f; }

  const int nch = causal ? ((qi0 >> 3) + 2) : (SEQ_T / 32);
  const int maskStart = causal ? (qi0 >> 3) : 0x7fffffff;

  __bf16* P = Plds + (threadIdx.x >> 5) * (16 * 32);
  const __bf16* vbh   = vt + ((size_t)(b * NH + h)) * HD * SEQ_T;
  const __bf16* kbase = kb + (size_t)b * SEQ_T * CH + h * HD;

  // prologue: K fragments for chunk 0 (col = key, K-dim = d)
  v16bf k00, k01, k10, k11;
  {
    const __bf16* kr0 = kbase;
    k00 = load_frag_b(kr0, CH);
    k01 = load_frag_b(kr0 + 32, CH);
    k10 = load_frag_b(kr0 + (size_t)16 * CH, CH);
    k11 = load_frag_b(kr0 + (size_t)16 * CH + 32, CH);
  }

#pragma unroll 2
  for (int ch = 0; ch < nch; ch++) {
    const int key0 = ch * 32;

    v8f s0 = {}, s1 = {};
    s0 = wmma_bf16(qa0, k00, s0);
    s0 = wmma_bf16(qa1, k01, s0);
    s1 = wmma_bf16(qa0, k10, s1);
    s1 = wmma_bf16(qa1, k11, s1);

    // issue V loads for this chunk early (consumed after the LDS round trip)
    v16bf vb0 = load_frag_b(vbh + (size_t)0  * SEQ_T + key0, SEQ_T);
    v16bf vb1 = load_frag_b(vbh + (size_t)16 * SEQ_T + key0, SEQ_T);
    v16bf vb2 = load_frag_b(vbh + (size_t)32 * SEQ_T + key0, SEQ_T);
    v16bf vb3 = load_frag_b(vbh + (size_t)48 * SEQ_T + key0, SEQ_T);

    // issue next chunk's K fragment loads (branchless clamp on last chunk)
    {
      const int keyn = (ch + 1 < nch) ? (key0 + 32) : key0;
      const __bf16* krn = kbase + (size_t)keyn * CH;
      k00 = load_frag_b(krn, CH);
      k01 = load_frag_b(krn + 32, CH);
      k10 = load_frag_b(krn + (size_t)16 * CH, CH);
      k11 = load_frag_b(krn + (size_t)16 * CH + 32, CH);
    }

    if (ch >= maskStart) {
#pragma unroll
      for (int r = 0; r < 8; r++) {
        const int lim = 4 * (qi0 + r + 8 * hf) + 3;
        if (key0 + r15 > lim)      s0[r] = -1e30f;
        if (key0 + 16 + r15 > lim) s1[r] = -1e30f;
      }
    }

    // ---- online softmax: rows live across the 16 lanes of each half ----
    float pm[8];
#pragma unroll
    for (int r = 0; r < 8; r++) pm[r] = fmaxf(s0[r], s1[r]);
#pragma unroll
    for (int m = 1; m <= 8; m <<= 1) {
#pragma unroll
      for (int r = 0; r < 8; r++) pm[r] = fmaxf(pm[r], __shfl_xor(pm[r], m, 32));
    }

    float p0[8], p1[8], rs[8], alpha[8];
#pragma unroll
    for (int r = 0; r < 8; r++) {
      const float mn = fmaxf(mrun[r], pm[r]);
      alpha[r] = __expf(mrun[r] - mn);
      mrun[r] = mn;
      p0[r] = __expf(s0[r] - mn);
      p1[r] = __expf(s1[r] - mn);
      rs[r] = p0[r] + p1[r];
    }
#pragma unroll
    for (int m = 1; m <= 8; m <<= 1) {
#pragma unroll
      for (int r = 0; r < 8; r++) rs[r] += __shfl_xor(rs[r], m, 32);
    }
#pragma unroll
    for (int r = 0; r < 8; r++) lrun[r] = lrun[r] * alpha[r] + rs[r];
#pragma unroll
    for (int r = 0; r < 8; r++) {
      y0[r] *= alpha[r]; y1[r] *= alpha[r]; y2[r] *= alpha[r]; y3[r] *= alpha[r];
    }

    // ---- C-layout -> A-layout for P via wave-private LDS ----
#pragma unroll
    for (int r = 0; r < 8; r++) {
      P[(r + 8 * hf) * 32 + r15]      = (__bf16)p0[r];
      P[(r + 8 * hf) * 32 + 16 + r15] = (__bf16)p1[r];
    }
    asm volatile("s_wait_dscnt 0" ::: "memory");
    v16bf pa = load_frag_a(P, 32);

    // ---- y += P * V ; vT[b,h,d,t] is a natural B operand (col=d, K=t) ----
    y0 = wmma_bf16(pa, vb0, y0);
    y1 = wmma_bf16(pa, vb1, y1);
    y2 = wmma_bf16(pa, vb2, y2);
    y3 = wmma_bf16(pa, vb3, y3);
  }

  float inv[8];
#pragma unroll
  for (int r = 0; r < 8; r++) inv[r] = 1.0f / lrun[r];
  __bf16* ybase = yb + ((size_t)b * SEQ_S + qi0) * CH + h * HD;
#pragma unroll
  for (int r = 0; r < 8; r++) {
    const size_t ro = (size_t)(r + 8 * hf) * CH;
    ybase[ro + 0  + r15] = (__bf16)(y0[r] * inv[r]);
    ybase[ro + 16 + r15] = (__bf16)(y1[r] * inv[r]);
    ybase[ro + 32 + r15] = (__bf16)(y2[r] * inv[r]);
    ybase[ro + 48 + r15] = (__bf16)(y3[r] * inv[r]);
  }
}

// ---------------------------------------------------------------------------
// Host launcher
// ---------------------------------------------------------------------------
extern "C" void kernel_launch(void* const* d_in, const int* in_sizes, int n_in,
                              void* d_out, int out_size, void* d_ws, size_t ws_size,
                              hipStream_t stream) {
  const float* x  = (const float*)d_in[0];
  const float* sx = (const float*)d_in[1];
  const float* Wq = (const float*)d_in[2];
  const float* Wk = (const float*)d_in[3];
  const float* Wv = (const float*)d_in[4];
  const float* Wc = (const float*)d_in[5];
  const float* qm = (const float*)d_in[6];
  const int* causal = (const int*)d_in[7];

  char* ws = (char*)d_ws;
  size_t off = 0;
  auto alloc = [&](size_t bytes) -> void* {
    void* p = ws + off;
    off += (bytes + 255) & ~(size_t)255;
    return p;
  };
  __bf16* xb   = (__bf16*)alloc((size_t)BATCH * SEQ_T * CH * 2);
  __bf16* sxb  = (__bf16*)alloc((size_t)BATCH * SEQ_S * CH * 2);
  __bf16* wqb  = (__bf16*)alloc((size_t)CH * CH * 2);
  __bf16* wkb  = (__bf16*)alloc((size_t)CH * CH * 2);
  __bf16* wvb  = (__bf16*)alloc((size_t)CH * CH * 2);
  __bf16* wcb  = (__bf16*)alloc((size_t)CH * CH * 2);
  __bf16* qbuf = (__bf16*)alloc((size_t)BATCH * SEQ_S * CH * 2);
  __bf16* kbuf = (__bf16*)alloc((size_t)BATCH * SEQ_T * CH * 2);
  __bf16* vtb  = (__bf16*)alloc((size_t)BATCH * SEQ_T * CH * 2);  // [B,H,D,T]
  __bf16* ybuf = (__bf16*)alloc((size_t)BATCH * SEQ_S * CH * 2);

  auto cvt = [&](const float* src, __bf16* dst, size_t n) {
    int n4 = (int)(n / 4);
    int blocks = (n4 + 255) / 256;
    if (blocks > 2048) blocks = 2048;
    cvt_f32_bf16<<<blocks, 256, 0, stream>>>(src, dst, n4);
  };
  cvt(x,  xb,  (size_t)BATCH * SEQ_T * CH);
  cvt(sx, sxb, (size_t)BATCH * SEQ_S * CH);
  cvt(Wq, wqb, (size_t)CH * CH);
  cvt(Wk, wkb, (size_t)CH * CH);
  cvt(Wv, wvb, (size_t)CH * CH);
  cvt(Wc, wcb, (size_t)CH * CH);

  // q = sx @ Wq^T, scaled by ln(T)*qm/sqrt(D)
  {
    const int M = BATCH * SEQ_S;
    const int waves = (M / 32) * (CH / 64);
    gemm_bf16_kernel<1><<<waves / 8, 256, 0, stream>>>(sxb, wqb, qbuf, qm, M, CH, CH);
  }
  // k = x @ Wk^T (row-major), v = x @ Wv^T (transposed per head)
  {
    const int M = BATCH * SEQ_T;
    const int waves = (M / 32) * (CH / 64);
    gemm_bf16_kernel<0><<<waves / 8, 256, 0, stream>>>(xb, wkb, kbuf, nullptr, M, CH, CH);
    gemm_bf16_kernel<2><<<waves / 8, 256, 0, stream>>>(xb, wvb, vtb,  nullptr, M, CH, CH);
  }
  // flash attention: 2048 wave-tiles, 8 waves/block
  attn_kernel<<<(BATCH * NH * (SEQ_S / 16)) / 8, 256, 0, stream>>>(qbuf, kbuf, vtb,
                                                                   ybuf, causal);
  // out = y @ Wc^T (f32)
  {
    const int M = BATCH * SEQ_S;
    const int waves = (M / 32) * (CH / 64);
    gemm_bf16_kernel<3><<<waves / 8, 256, 0, stream>>>(ybuf, wcb, d_out, nullptr, M, CH, CH);
  }
}
